// Attention_14035953123996
// MI455X (gfx1250) — matrix operand order repaired and checked by reading, not compiled, not run on hardware
//
#include <hip/hip_runtime.h>
#include <math.h>

#define Bb 64
#define Tt 2048
#define DQd 1024
#define Aa 128
#define Ee 512
#define NFf 32
#define KKc 31
#define PADc 15
#define CHUNKS 16
#define TCH (Tt / CHUNKS) /* 128 */

#define WM_STRIDE 520  /* bf16 elems per staged W_memory row (512 + 8 pad) */
#define WL_STRIDE 40   /* bf16 elems per staged W_loc / locf row (32 + 8 pad) */

typedef __attribute__((ext_vector_type(16))) __bf16   v16bf;
typedef __attribute__((ext_vector_type(8)))  float    v8f;
typedef __attribute__((ext_vector_type(8)))  int      v8i;
typedef __attribute__((ext_vector_type(4)))  float    v4f;
typedef __attribute__((ext_vector_type(4)))  unsigned v4u;

// Pack two f32 into two bf16 (round-to-nearest) with v_perm_b32.
__device__ __forceinline__ unsigned pack_bf16(float a, float b) {
  unsigned ua = __builtin_bit_cast(unsigned, a) + 0x8000u;
  unsigned ub = __builtin_bit_cast(unsigned, b) + 0x8000u;
  return __builtin_amdgcn_perm(ub, ua, 0x07060302u);  // { b.hi16, a.hi16 }
}

__device__ __forceinline__ __bf16 f2bf(float f) {
  union { float f; unsigned u; } x; x.f = f;
  unsigned r = x.u + 0x7FFFu + ((x.u >> 16) & 1u);   // RNE
  unsigned short h = (unsigned short)(r >> 16);
  __bf16 b;
  __builtin_memcpy(&b, &h, sizeof(b));
  return b;
}

__device__ __forceinline__ float fast_tanh(float x) {
  x = fminf(fmaxf(x, -15.f), 15.f);
  const float e2 = __expf(2.f * x);
  return (e2 - 1.f) / (e2 + 1.f);
}

// Load + convert one 16-bit A fragment chunk (16 f32 -> v16bf) from p.
__device__ __forceinline__ v16bf load_a_frag(const float* p) {
  const v4f* ap0 = (const v4f*)(p);
  const v4f* ap1 = (const v4f*)(p + 16);
  const v4f a0 = ap0[0], a1 = ap0[1], a2 = ap1[0], a3 = ap1[1];
  v8i ai;
  ai[0] = (int)pack_bf16(a0[0], a0[1]);
  ai[1] = (int)pack_bf16(a0[2], a0[3]);
  ai[2] = (int)pack_bf16(a1[0], a1[1]);
  ai[3] = (int)pack_bf16(a1[2], a1[3]);
  ai[4] = (int)pack_bf16(a2[0], a2[1]);
  ai[5] = (int)pack_bf16(a2[2], a2[3]);
  ai[6] = (int)pack_bf16(a3[0], a3[1]);
  ai[7] = (int)pack_bf16(a3[2], a3[3]);
  return __builtin_bit_cast(v16bf, ai);
}

__device__ __forceinline__ v16bf combine_b(v4u lo, v4u hi) {
  v8i bi;
  bi[0] = (int)lo[0]; bi[1] = (int)lo[1]; bi[2] = (int)lo[2]; bi[3] = (int)lo[3];
  bi[4] = (int)hi[0]; bi[5] = (int)hi[1]; bi[6] = (int)hi[2]; bi[7] = (int)hi[3];
  return __builtin_bit_cast(v16bf, bi);
}

// Pass 1: one block per (batch, T-chunk). 256 threads = 8 waves.
__global__ __launch_bounds__(256)
void attn_pass1(const float* __restrict__ query,
                const float* __restrict__ memory,
                const float* __restrict__ awcat,
                const float* __restrict__ Wq,
                const float* __restrict__ Wm,
                const float* __restrict__ convw,
                const float* __restrict__ Wloc,
                const float* __restrict__ vvec,
                float* __restrict__ out_w,   // [B][T] raw energies
                float* __restrict__ ws_ms,   // [B][CHUNKS][2] (m, s)
                float* __restrict__ ws_c)    // [B][CHUNKS][E] partial contexts
{
  extern __shared__ char smem_raw[];
  __bf16* wm_s   = (__bf16*)(smem_raw);            // [Aa][WM_STRIDE] 133120 B
  __bf16* wl_s   = (__bf16*)(smem_raw + 133120);   // [Aa][WL_STRIDE]  10240 B
  __bf16* locf_s = (__bf16*)(smem_raw + 143360);   // [TCH][WL_STRIDE] 10240 B
  float*  q_s    = (float*)(smem_raw + 153600);    // [Aa]               512 B
  float*  e_s    = (float*)(smem_raw + 154112);    // [TCH]              512 B
  float*  ms_s   = (float*)(smem_raw + 154624);    // [2]                  8 B
  // total 154632 B -> ~152 KB => 2 blocks per WGP

  const int blk   = blockIdx.x;
  const int b     = blk / CHUNKS;
  const int chunk = blk % CHUNKS;
  const int tid   = threadIdx.x;
  const int wave  = tid >> 5;
  const int lane  = tid & 31;
  const int half  = lane >> 4;   // 0: lanes 0-15, 1: lanes 16-31
  const int ln    = lane & 15;
  const int tbase = chunk * TCH;

  // ---- stage weights into LDS (bf16, padded rows) ----
  {
    unsigned* wm_u = (unsigned*)wm_s;  // row stride WM_STRIDE/2 = 260
    for (int i = tid; i < Aa * (Ee / 2); i += 256) {
      const int row = i >> 8;          // 256 pairs per row
      const int cp  = i & 255;
      const float* src = Wm + (size_t)row * Ee + cp * 2;
      wm_u[row * (WM_STRIDE / 2) + cp] = pack_bf16(src[0], src[1]);
    }
    unsigned* wl_u = (unsigned*)wl_s;  // row stride WL_STRIDE/2 = 20
    for (int i = tid; i < Aa * (NFf / 2); i += 256) {
      const int row = i >> 4;          // 16 pairs per row
      const int cp  = i & 15;
      const float* src = Wloc + (size_t)row * NFf + cp * 2;
      wl_u[row * (WL_STRIDE / 2) + cp] = pack_bf16(src[0], src[1]);
    }
  }

  // ---- q projection: 2 threads (adjacent lanes) per output a ----
  {
    const int a2 = tid >> 1, h = tid & 1;
    const float* qp = query + (size_t)b * DQd + h * (DQd / 2);
    const float* wq = Wq + (size_t)a2 * DQd + h * (DQd / 2);
    float s = 0.f;
    #pragma unroll 8
    for (int d = 0; d < DQd / 2; ++d) s += qp[d] * wq[d];
    s += __shfl_xor(s, 1, 32);
    if (h == 0) q_s[a2] = s;
  }

  // ---- location conv: thread owns filter f for 16 consecutive t ----
  // sliding window of aw kept in registers; weights read per-tap (L2-hot)
  {
    const int f  = tid & 31;
    const int t0 = (tid >> 5) * 16;       // 8 groups of 16 t
    const float* aw0 = awcat + ((size_t)b * 2 + 0) * Tt;
    const float* aw1 = awcat + ((size_t)b * 2 + 1) * Tt;
    const float* w0  = convw + (f * 2 + 0) * KKc;
    const float* w1  = convw + (f * 2 + 1) * KKc;
    float buf0[46], buf1[46];
    #pragma unroll
    for (int i = 0; i < 46; ++i) {
      const int src = tbase + t0 + i - PADc;
      const bool ok = (src >= 0) && (src < Tt);
      buf0[i] = ok ? aw0[src] : 0.f;
      buf1[i] = ok ? aw1[src] : 0.f;
    }
    float s[16];
    #pragma unroll
    for (int j = 0; j < 16; ++j) s[j] = 0.f;
    #pragma unroll
    for (int k = 0; k < KKc; ++k) {
      const float wk0 = w0[k], wk1 = w1[k];
      #pragma unroll
      for (int j = 0; j < 16; ++j) s[j] += buf0[j + k] * wk0 + buf1[j + k] * wk1;
    }
    #pragma unroll
    for (int j = 0; j < 16; ++j) locf_s[(t0 + j) * WL_STRIDE + f] = f2bf(s[j]);
  }
  __syncthreads();

  // ---- WMMA GEMM: acc[16 t x 128 a] = mem_chunk x Wm^T (+ locf x Wloc^T) ----
  v8f acc[8];
  #pragma unroll
  for (int tn = 0; tn < 8; ++tn) acc[tn] = {};

  const int    trow = wave * 16 + ln;   // A-frag row (same for both lane halves)
  const float* arow = memory + ((size_t)b * Tt + tbase + trow) * Ee;

  v16bf afrag = load_a_frag(arow + half * 8);
  for (int kk = 0; kk < Ee / 32; ++kk) {
    const int k0 = kk * 32;
    // preload all 8 B fragments for this k step (ds loads pipeline in-order)
    v4u blo[8], bhi[8];
    #pragma unroll
    for (int tn = 0; tn < 8; ++tn) {
      const v4u* wp =
          (const v4u*)(wm_s + (size_t)(tn * 16 + ln) * WM_STRIDE + k0 + half * 8);
      blo[tn] = wp[0];
      bhi[tn] = wp[2];
    }
    // software-pipeline next A fragment + prefetch further ahead
    if (kk + 2 < Ee / 32)
      __builtin_prefetch(arow + (kk + 2) * 32 + half * 8, 0, 3);
    v16bf anext = afrag;
    if (kk + 1 < Ee / 32) anext = load_a_frag(arow + (kk + 1) * 32 + half * 8);
    #pragma unroll
    for (int tn = 0; tn < 8; ++tn) {
      acc[tn] = __builtin_amdgcn_wmma_f32_16x16x32_bf16(
          false, afrag, false, combine_b(blo[tn], bhi[tn]), (short)0, acc[tn],
          false, false);
    }
    afrag = anext;
  }

  // location term: one extra K=32 step (K = NF) into the same accumulators
  {
    const v4u* lp0 = (const v4u*)(locf_s + (size_t)trow * WL_STRIDE + half * 8);
    const v4u* lp1 = (const v4u*)(locf_s + (size_t)trow * WL_STRIDE + half * 8 + 16);
    const v16bf lfrag = combine_b(lp0[0], lp1[0]);
    v4u blo[8], bhi[8];
    #pragma unroll
    for (int tn = 0; tn < 8; ++tn) {
      const v4u* wp =
          (const v4u*)(wl_s + (size_t)(tn * 16 + ln) * WL_STRIDE + half * 8);
      blo[tn] = wp[0];
      bhi[tn] = wp[2];
    }
    #pragma unroll
    for (int tn = 0; tn < 8; ++tn) {
      acc[tn] = __builtin_amdgcn_wmma_f32_16x16x32_bf16(
          false, lfrag, false, combine_b(blo[tn], bhi[tn]), (short)0, acc[tn],
          false, false);
    }
  }

  // ---- energies: e_t = sum_a v[a]*tanh(q[a] + acc[t][a]) ----
  // D layout: VGPR r -> (M=r, N=lane) lanes 0-15, (M=8+r, N=lane-16) lanes 16-31
  float er[8];
  #pragma unroll
  for (int r = 0; r < 8; ++r) er[r] = 0.f;
  #pragma unroll
  for (int tn = 0; tn < 8; ++tn) {
    const int a = tn * 16 + ln;
    const float qa = q_s[a];
    const float va = vvec[a];
    #pragma unroll
    for (int r = 0; r < 8; ++r) er[r] += va * fast_tanh(acc[tn][r] + qa);
  }
  #pragma unroll
  for (int r = 0; r < 8; ++r) {
    er[r] += __shfl_xor(er[r], 1, 32);
    er[r] += __shfl_xor(er[r], 2, 32);
    er[r] += __shfl_xor(er[r], 4, 32);
    er[r] += __shfl_xor(er[r], 8, 32);
  }
  if (ln == 0) {
    #pragma unroll
    for (int r = 0; r < 8; ++r) {
      const int tl = wave * 16 + half * 8 + r;
      e_s[tl] = er[r];
      out_w[(size_t)b * Tt + tbase + tl] = er[r];   // raw; pass2 normalizes
    }
  }
  __syncthreads();

  // ---- chunk-local softmax stats (wave 0) ----
  if (wave == 0) {
    float m = -1e30f;
    #pragma unroll
    for (int i = 0; i < 4; ++i) m = fmaxf(m, e_s[lane + 32 * i]);
    #pragma unroll
    for (int msk = 16; msk >= 1; msk >>= 1) m = fmaxf(m, __shfl_xor(m, msk, 32));
    float s = 0.f;
    #pragma unroll
    for (int i = 0; i < 4; ++i) s += __expf(e_s[lane + 32 * i] - m);
    #pragma unroll
    for (int msk = 16; msk >= 1; msk >>= 1) s += __shfl_xor(s, msk, 32);
    if (lane == 0) {
      ms_s[0] = m; ms_s[1] = s;
      float* msp = ws_ms + ((size_t)b * CHUNKS + chunk) * 2;
      msp[0] = m; msp[1] = s;
    }
  }
  __syncthreads();

  // ---- partial context: c[e] = sum_t exp(e_t - m) * mem[t][e] (chunk L2-hot) ----
  const float m = ms_s[0];
  float c0 = 0.f, c1 = 0.f;
  const float* mb = memory + ((size_t)b * Tt + tbase) * Ee;
  for (int t = 0; t < TCH; ++t) {
    const float p = __expf(e_s[t] - m);
    c0 += p * mb[(size_t)t * Ee + tid];
    c1 += p * mb[(size_t)t * Ee + tid + 256];
  }
  float* cp = ws_c + ((size_t)b * CHUNKS + chunk) * Ee;
  cp[tid]       = c0;
  cp[tid + 256] = c1;
}

// Pass 2: combine per-chunk online-softmax partials; normalize weights, emit context.
__global__ __launch_bounds__(256)
void attn_pass2(const float* __restrict__ ws_ms,
                const float* __restrict__ ws_c,
                float* __restrict__ out_ctx,  // [B][E]
                float* __restrict__ out_w)    // [B][T] raw -> normalized in-place
{
  const int b   = blockIdx.x;
  const int tid = threadIdx.x;
  float mi[CHUNKS], si[CHUNKS], sc[CHUNKS];
  #pragma unroll
  for (int i = 0; i < CHUNKS; ++i) {
    mi[i] = ws_ms[((size_t)b * CHUNKS + i) * 2 + 0];
    si[i] = ws_ms[((size_t)b * CHUNKS + i) * 2 + 1];
  }
  float m = -1e30f;
  #pragma unroll
  for (int i = 0; i < CHUNKS; ++i) m = fmaxf(m, mi[i]);
  float s = 0.f;
  #pragma unroll
  for (int i = 0; i < CHUNKS; ++i) { sc[i] = __expf(mi[i] - m); s += si[i] * sc[i]; }
  const float inv_s = 1.0f / s;

  for (int e = tid; e < Ee; e += 256) {
    float acc = 0.f;
    #pragma unroll
    for (int i = 0; i < CHUNKS; ++i)
      acc += ws_c[((size_t)b * CHUNKS + i) * Ee + e] * sc[i];
    out_ctx[(size_t)b * Ee + e] = acc * inv_s;
  }
  for (int t = tid; t < Tt; t += 256) {
    const float eraw = out_w[(size_t)b * Tt + t];
    out_w[(size_t)b * Tt + t] = __expf(eraw - m) * inv_s;
  }
}

extern "C" void kernel_launch(void* const* d_in, const int* in_sizes, int n_in,
                              void* d_out, int out_size, void* d_ws, size_t ws_size,
                              hipStream_t stream) {
  (void)in_sizes; (void)n_in; (void)out_size; (void)ws_size;
  const float* query  = (const float*)d_in[0];
  const float* memory = (const float*)d_in[1];
  const float* awcat  = (const float*)d_in[2];
  const float* Wq     = (const float*)d_in[3];
  const float* Wm     = (const float*)d_in[4];
  const float* convw  = (const float*)d_in[5];
  const float* Wloc   = (const float*)d_in[6];
  const float* vvec   = (const float*)d_in[7];

  float* out     = (float*)d_out;
  float* out_ctx = out;                    // [B][E]
  float* out_w   = out + (size_t)Bb * Ee;  // [B][T]

  float* ws_ms = (float*)d_ws;                        // [B][CHUNKS][2]
  float* ws_c  = ws_ms + (size_t)Bb * CHUNKS * 2;     // [B][CHUNKS][E]

  const size_t smem_bytes = 154632;
  attn_pass1<<<Bb * CHUNKS, 256, smem_bytes, stream>>>(query, memory, awcat,
                                                       Wq, Wm, convw, Wloc, vvec,
                                                       out_w, ws_ms, ws_c);
  attn_pass2<<<Bb, 256, 0, stream>>>(ws_ms, ws_c, out_ctx, out_w);
}
